// BiEncoder_77309411328346
// MI455X (gfx1250) — matrix operand order repaired and checked by reading, not compiled, not run on hardware
//
#include <hip/hip_runtime.h>
#include <hip/hip_bf16.h>
#include <cstdint>
#include <cstddef>

// Problem constants (match reference)
#define cB   256
#define cT   64
#define cF   2048
#define cP   512
#define cH   512
#define cH4  2048
#define cBT  (cB * cT)

typedef __attribute__((ext_vector_type(16))) __bf16 v16bf;
typedef __attribute__((ext_vector_type(8)))  __bf16 v8bf;
typedef __attribute__((ext_vector_type(8)))  float  v8f;

// ---------------------------------------------------------------------------
// WMMA fragment loaders (layouts per CDNA5 ISA 7.12.2, 16-bit A 16x32 / B 32x16)
// A (MxK row-major): lane L holds row rowBase+(L&15); elements 0..7 -> k0+g*8..+7,
//                    elements 8..15 -> k0+16+g*8..+7, g = L>>4.
// B (stored as NxK row-major weights, i.e. B^T): lane L holds col rowBase+(L&15);
//                    elements 0..7 -> k0+g*8..+7, elements 8..15 -> k0+16+g*8..+7.
// ---------------------------------------------------------------------------
__device__ __forceinline__ v16bf load_frag_a_bf16(const __bf16* __restrict__ base,
                                                  int ld, int rowBase, int k0) {
  const int lane = threadIdx.x & 31;
  const int g = lane >> 4;
  const __bf16* p = base + (size_t)(rowBase + (lane & 15)) * ld + k0 + g * 8;
  v8bf lo = *(const v8bf*)(p);
  v8bf hi = *(const v8bf*)(p + 16);
  return __builtin_shufflevector(lo, hi, 0,1,2,3,4,5,6,7,8,9,10,11,12,13,14,15);
}

__device__ __forceinline__ v16bf load_frag_b_bf16(const __bf16* __restrict__ base,
                                                  int ld, int rowBase, int k0) {
  const int lane = threadIdx.x & 31;
  const int g = lane >> 4;
  const __bf16* p = base + (size_t)(rowBase + (lane & 15)) * ld + k0 + g * 8;
  v8bf lo = *(const v8bf*)(p);
  v8bf hi = *(const v8bf*)(p + 16);
  return __builtin_shufflevector(lo, hi, 0,1,2,3,4,5,6,7,8,9,10,11,12,13,14,15);
}

// A fragment directly from fp32 source (projection input), converted in-register.
__device__ __forceinline__ v16bf load_frag_a_f32(const float* __restrict__ base,
                                                 int ld, int rowBase, int k0) {
  const int lane = threadIdx.x & 31;
  const int g = lane >> 4;
  const float* p = base + (size_t)(rowBase + (lane & 15)) * ld + k0 + g * 8;
  v16bf r;
#pragma unroll
  for (int i = 0; i < 8; ++i) {
    r[i]     = (__bf16)p[i];
    r[i + 8] = (__bf16)p[16 + i];
  }
  return r;
}

// ---------------------------------------------------------------------------
// Big parallel GEMM:  C_bf16[M,N] = A[M,K] * W[N,K]^T + bias[N]
// Block = 256 threads = 8 waves as 4(M) x 2(N); block tile 128x128;
// each wave: 32x64 tile = 2x4 v_wmma_f32_16x16x32_bf16 accumulators.
// Grid: (M/128, N/128). All call sites have divisible M,N,K.
// ---------------------------------------------------------------------------
template <bool A_F32>
__global__ __launch_bounds__(256) void wmma_gemm_kernel(
    const float*  __restrict__ Af32,  // fp32 A (A_F32)
    const __bf16* __restrict__ Abf,   // bf16 A (!A_F32)
    int lda,
    const __bf16* __restrict__ W,     // N x K row-major (ld = K)
    const float*  __restrict__ bias,  // len N
    __bf16* __restrict__ outBF,       // bf16 out
    int ldc, int K)
{
  const int lane = threadIdx.x & 31;
  const int w    = threadIdx.x >> 5;
  const int m0   = blockIdx.x * 128 + (w >> 1) * 32;
  const int n0   = blockIdx.y * 128 + (w & 1)  * 64;

  v8f acc[2][4] = {};

  for (int kk = 0; kk < K; kk += 32) {
    v16bf a[2];
    if (A_F32) {
      a[0] = load_frag_a_f32(Af32, lda, m0,      kk);
      a[1] = load_frag_a_f32(Af32, lda, m0 + 16, kk);
    } else {
      a[0] = load_frag_a_bf16(Abf, lda, m0,      kk);
      a[1] = load_frag_a_bf16(Abf, lda, m0 + 16, kk);
    }
    v16bf b[4];
#pragma unroll
    for (int j = 0; j < 4; ++j)
      b[j] = load_frag_b_bf16(W, K, n0 + j * 16, kk);

#pragma unroll
    for (int i = 0; i < 2; ++i)
#pragma unroll
      for (int j = 0; j < 4; ++j)
        acc[i][j] = __builtin_amdgcn_wmma_f32_16x16x32_bf16(
            false, a[i], false, b[j], (short)0, acc[i][j], false, false);
  }

  // C/D layout: lane holds col n0+(lane&15); VGPR r holds row m0+(lane>>4)*8+r.
  const int rb = (lane >> 4) * 8;
  const int cn = lane & 15;
#pragma unroll
  for (int j = 0; j < 4; ++j) {
    const int col = n0 + j * 16 + cn;
    const float bv = bias[col];
#pragma unroll
    for (int i = 0; i < 2; ++i) {
#pragma unroll
      for (int r = 0; r < 8; ++r) {
        const int row = m0 + i * 16 + rb + r;
        outBF[(size_t)row * ldc + col] = (__bf16)(acc[i][j][r] + bv);
      }
    }
  }
}

// ---------------------------------------------------------------------------
// Fused recurrent step: for 16 batch rows, compute g = G[:,t] + h @ W_hh^T
// over ALL 4H columns, stage g in LDS, then apply LSTM gates in-block.
// Block = 512 threads (16 waves); wave w covers cols [w*128, w*128+128)
// as 8 WMMA accumulators. Grid: (B/16, 1, 2) — z selects LSTM direction.
// Dynamic LDS: 16 x (cH4+4) fp32 (padding kills bank conflicts).
// Each block reads/writes only its own 16 h/c rows: no cross-block hazard.
// ---------------------------------------------------------------------------
__global__ __launch_bounds__(512) void lstm_step_fused_kernel(
    __bf16* __restrict__ h1, const __bf16* __restrict__ W1, const __bf16* __restrict__ G1,
    float* __restrict__ c1, float* __restrict__ out1, int t1,
    __bf16* __restrict__ h2, const __bf16* __restrict__ W2, const __bf16* __restrict__ G2,
    float* __restrict__ c2, float* __restrict__ out2, int t2)
{
  extern __shared__ float gsh[];
  constexpr int GROW = cH4 + 4;

  __bf16* h; const __bf16 *W, *G; float *c, *out; int t;
  if (blockIdx.z == 0) { h = h1; W = W1; G = G1; c = c1; out = out1; t = t1; }
  else                 { h = h2; W = W2; G = G2; c = c2; out = out2; t = t2; }

  const int lane = threadIdx.x & 31;
  const int w    = threadIdx.x >> 5;      // 0..15
  const int m0   = blockIdx.x * 16;       // batch row base
  const int n0   = w * 128;               // column base (wave tile 16x128)

  v8f acc[8] = {};
  for (int kk = 0; kk < cH; kk += 32) {
    v16bf a = load_frag_a_bf16(h, cH, m0, kk);
#pragma unroll
    for (int j = 0; j < 8; ++j) {
      v16bf b = load_frag_b_bf16(W, cH, n0 + j * 16, kk);
      acc[j] = __builtin_amdgcn_wmma_f32_16x16x32_bf16(
          false, a, false, b, (short)0, acc[j], false, false);
    }
  }

  // Epilogue: g = acc + G[b, t, col] -> LDS
  const int rb = (lane >> 4) * 8;
  const int cn = lane & 15;
  const size_t gstride = (size_t)cT * cH4;   // G row stride (rows = batch)
#pragma unroll
  for (int j = 0; j < 8; ++j) {
    const int col = n0 + j * 16 + cn;
#pragma unroll
    for (int r = 0; r < 8; ++r) {
      const int row = rb + r;               // 0..15 within tile
      const float v = acc[j][r] +
          (float)G[(size_t)(m0 + row) * gstride + (size_t)t * cH4 + col];
      gsh[row * GROW + col] = v;
    }
  }
  __syncthreads();

  // Gates: 16 rows x cH hidden units = 8192 elements over 512 threads.
  for (int e = threadIdx.x; e < 16 * cH; e += 512) {
    const int r = e >> 9;                   // / cH
    const int n = e & (cH - 1);
    const float* gr = gsh + r * GROW;
    const float gi = gr[n];
    const float gf = gr[cH + n];
    const float gg = gr[2 * cH + n];
    const float go = gr[3 * cH + n];

    const float si = 1.0f / (1.0f + __expf(-gi));
    const float sf = 1.0f / (1.0f + __expf(-gf));
    const float so = 1.0f / (1.0f + __expf(-go));

    const int b = m0 + r;
    float cc = c[(size_t)b * cH + n];
    cc = sf * cc + si * tanhf(gg);
    const float hv = so * tanhf(cc);

    c[(size_t)b * cH + n] = cc;
    h[(size_t)b * cH + n] = (__bf16)hv;
    out[(size_t)b * (cT * cH) + (size_t)t * cH + n] = hv;
  }
}

// ---------------------------------------------------------------------------
// Elementwise helpers
// ---------------------------------------------------------------------------
__global__ __launch_bounds__(256) void f32_to_bf16_kernel(const float* __restrict__ in,
                                                          __bf16* __restrict__ out, int n) {
  int i = blockIdx.x * 256 + threadIdx.x;
  if (i < n) out[i] = (__bf16)in[i];
}

__global__ __launch_bounds__(256) void bias_combine_kernel(
    const float* __restrict__ bi1, const float* __restrict__ bh1, float* __restrict__ b1,
    const float* __restrict__ bi2, const float* __restrict__ bh2, float* __restrict__ b2, int n) {
  int i = blockIdx.x * 256 + threadIdx.x;
  if (i < n) { b1[i] = bi1[i] + bh1[i]; b2[i] = bi2[i] + bh2[i]; }
}

__global__ __launch_bounds__(256) void zero_state_kernel(float* c1, float* c2,
                                                         __bf16* h1, __bf16* h2, int n) {
  int i = blockIdx.x * 256 + threadIdx.x;
  if (i < n) { c1[i] = 0.0f; c2[i] = 0.0f; h1[i] = (__bf16)0.0f; h2[i] = (__bf16)0.0f; }
}

// ---------------------------------------------------------------------------
// Host orchestration
// ---------------------------------------------------------------------------
extern "C" void kernel_launch(void* const* d_in, const int* in_sizes, int n_in,
                              void* d_out, int out_size, void* d_ws, size_t ws_size,
                              hipStream_t stream) {
  (void)in_sizes; (void)n_in; (void)out_size;

  const float* X     = (const float*)d_in[0];   // (B,T,F)
  const float* W_e   = (const float*)d_in[1];   // (P,F)
  const float* b_e   = (const float*)d_in[2];   // (P)
  const float* W_ih1 = (const float*)d_in[3];   // (4H,P)
  const float* W_hh1 = (const float*)d_in[4];   // (4H,H)
  const float* b_ih1 = (const float*)d_in[5];
  const float* b_hh1 = (const float*)d_in[6];
  const float* W_ih2 = (const float*)d_in[7];
  const float* W_hh2 = (const float*)d_in[8];
  const float* b_ih2 = (const float*)d_in[9];
  const float* b_hh2 = (const float*)d_in[10];

  char* ws = (char*)d_ws;
  size_t off = 0;
  auto alloc = [&](size_t bytes) -> void* {
    void* p = (void*)(ws + off);
    off += (bytes + 255) & ~(size_t)255;
    return p;
  };

  __bf16* We_bf   = (__bf16*)alloc((size_t)cP  * cF * 2);
  __bf16* Wih1_bf = (__bf16*)alloc((size_t)cH4 * cP * 2);
  __bf16* Whh1_bf = (__bf16*)alloc((size_t)cH4 * cH * 2);
  __bf16* Wih2_bf = (__bf16*)alloc((size_t)cH4 * cP * 2);
  __bf16* Whh2_bf = (__bf16*)alloc((size_t)cH4 * cH * 2);
  float*  b1      = (float*) alloc((size_t)cH4 * 4);
  float*  b2      = (float*) alloc((size_t)cH4 * 4);
  __bf16* Vbf     = (__bf16*)alloc((size_t)cBT * cP * 2);
  __bf16* G1      = (__bf16*)alloc((size_t)cBT * cH4 * 2);
  __bf16* G2      = (__bf16*)alloc((size_t)cBT * cH4 * 2);
  float*  c1      = (float*) alloc((size_t)cB * cH * 4);
  float*  c2      = (float*) alloc((size_t)cB * cH * 4);
  __bf16* h1bf    = (__bf16*)alloc((size_t)cB * cH * 2);
  __bf16* h2bf    = (__bf16*)alloc((size_t)cB * cH * 2);
  if (off > ws_size) return;  // workspace too small: bail (no-op)

  float* out1 = (float*)d_out;
  float* out2 = out1 + (size_t)cB * cT * cH;

  // Phase 0: weight conversion, bias fold, state init
  const int NW = cP * cF;  // == cH4*cP == cH4*cH == 1048576
  f32_to_bf16_kernel<<<(NW + 255) / 256, 256, 0, stream>>>(W_e,   We_bf,   NW);
  f32_to_bf16_kernel<<<(NW + 255) / 256, 256, 0, stream>>>(W_ih1, Wih1_bf, NW);
  f32_to_bf16_kernel<<<(NW + 255) / 256, 256, 0, stream>>>(W_hh1, Whh1_bf, NW);
  f32_to_bf16_kernel<<<(NW + 255) / 256, 256, 0, stream>>>(W_ih2, Wih2_bf, NW);
  f32_to_bf16_kernel<<<(NW + 255) / 256, 256, 0, stream>>>(W_hh2, Whh2_bf, NW);
  bias_combine_kernel<<<(cH4 + 255) / 256, 256, 0, stream>>>(b_ih1, b_hh1, b1,
                                                             b_ih2, b_hh2, b2, cH4);
  zero_state_kernel<<<(cB * cH + 255) / 256, 256, 0, stream>>>(c1, c2, h1bf, h2bf, cB * cH);

  // Phase 1: projection V = X @ W_e^T + b_e  (fp32 A converted in-register)
  {
    dim3 grid(cBT / 128, cP / 128);
    wmma_gemm_kernel<true><<<grid, 256, 0, stream>>>(
        X, nullptr, cF, We_bf, b_e, Vbf, cP, cF);
  }

  // Phase 2: gate pre-activations G = V @ W_ih^T + (b_ih + b_hh), stored bf16
  {
    dim3 grid(cBT / 128, cH4 / 128);
    wmma_gemm_kernel<false><<<grid, 256, 0, stream>>>(
        nullptr, Vbf, cP, Wih1_bf, b1, G1, cH4, cP);
    wmma_gemm_kernel<false><<<grid, 256, 0, stream>>>(
        nullptr, Vbf, cP, Wih2_bf, b2, G2, cH4, cP);
  }

  // Phase 3: sequential recurrence, ONE fused launch per step (both LSTMs).
  // lstm1 walks t = s (forward), lstm2 walks t = T-1-s (reverse).
  const dim3 sgrid(cB / 16, 1, 2);
  const size_t lds_bytes = (size_t)16 * (cH4 + 4) * sizeof(float);
  for (int s = 0; s < cT; ++s) {
    const int t2 = cT - 1 - s;
    lstm_step_fused_kernel<<<sgrid, 512, lds_bytes, stream>>>(
        h1bf, Whh1_bf, G1, c1, out1, s,
        h2bf, Whh2_bf, G2, c2, out2, t2);
  }
}
